// BertSelfAttention_70574902608134
// MI455X (gfx1250) — matrix operand order repaired and checked by reading, not compiled, not run on hardware
//
#include <hip/hip_runtime.h>
#include <math.h>

// ---------------------------------------------------------------------------
// BERT self-attention with block mask, MI455X (gfx1250), bf16 WMMA pipeline.
// D = A x B + C via v_wmma_f32_16x16x32_bf16 (wave32).
// ---------------------------------------------------------------------------

typedef __bf16 bf16;
typedef __attribute__((ext_vector_type(16))) __bf16 v16bf;
typedef __attribute__((ext_vector_type(8)))  __bf16 v8bf;
typedef __attribute__((ext_vector_type(8)))  float  v8f;
typedef __attribute__((ext_vector_type(4)))  float  v4f;

#define WMMA_BF16(A, B, C) \
  __builtin_amdgcn_wmma_f32_16x16x32_bf16(false, (A), false, (B), (short)0, (C), false, false)

// Problem constants (match setup_inputs()).
constexpr int Bn  = 8;
constexpr int Ln  = 64;     // signal_length
constexpr int CDD = 20;     // cdd_size
constexpr int Tn  = 128;    // term_num
constexpr int Dn  = 768;
constexpr int Hn  = 12;
constexpr int Dh  = 64;     // Dn / Hn
constexpr int AF  = CDD * Ln;   // 1280
constexpr int Sn  = AF + Tn;    // 1408
constexpr int Mn  = Bn * Sn;    // 11264

// A-operand 16x32 bf16 fragment: per-lane two contiguous 8-elt chunks.
__device__ inline v16bf make_a(const bf16* p0, const bf16* p1) {
  v8bf lo = *(const v8bf*)p0;
  v8bf hi = *(const v8bf*)p1;
  v16bf a;
#pragma unroll
  for (int e = 0; e < 8; ++e) { a[e] = lo[e]; a[e + 8] = hi[e]; }
  return a;
}

// B-operand 32x16 bf16 fragment: per-lane two 8-elt K chunks at p and p+16.
__device__ inline v16bf make_b(const bf16* p) {
  v8bf lo = *(const v8bf*)p;
  v8bf hi = *(const v8bf*)(p + 16);
  v16bf b;
#pragma unroll
  for (int e = 0; e < 8; ++e) { b[e] = lo[e]; b[e + 8] = hi[e]; }
  return b;
}

// ---------------------------------------------------------------------------
// fp32 -> bf16 conversion (one-time; bandwidth-trivial at 23.3 TB/s)
// ---------------------------------------------------------------------------
__global__ void f32_to_bf16(const float* __restrict__ src, bf16* __restrict__ dst, int n) {
  int i = (blockIdx.x * blockDim.x + threadIdx.x) * 8;
  if (i + 8 <= n) {
    v4f x0 = *(const v4f*)(src + i);
    v4f x1 = *(const v4f*)(src + i + 4);
    v8bf y;
#pragma unroll
    for (int e = 0; e < 4; ++e) { y[e] = (bf16)x0[e]; y[e + 4] = (bf16)x1[e]; }
    *(v8bf*)(dst + i) = y;
  } else {
    for (int k = i; k < n; ++k) dst[k] = (bf16)src[k];
  }
}

// ---------------------------------------------------------------------------
// Q/K projection: OUT[m][n] = sum_k hidden[m][k] * W[n][k] + bias[n]
// Transposed WMMA orientation (wmmaM = n, wmmaN = m) so D-fragments give
// contiguous 16B bf16 stores into row-major (B,S,D).
// Block: 8 waves = 2(n) x 4(m), tile 128n x 256m. Wave tile 64x64.
// ---------------------------------------------------------------------------
__global__ __launch_bounds__(256) void proj_qk(
    const bf16* __restrict__ hBF,
    const bf16* __restrict__ wqB, const float* __restrict__ bq,
    const bf16* __restrict__ wkB, const float* __restrict__ bk,
    bf16* __restrict__ Qbf, bf16* __restrict__ Kbf) {
  const bf16*  W    = (blockIdx.z == 0) ? wqB : wkB;
  const float* bias = (blockIdx.z == 0) ? bq : bk;
  bf16*        out  = (blockIdx.z == 0) ? Qbf : Kbf;

  int lane = threadIdx.x & 31;
  int w    = threadIdx.x >> 5;
  int wn = w & 1, wm = w >> 1;
  int nbase = blockIdx.y * 128 + wn * 64;
  int mbase = blockIdx.x * 256 + wm * 64;
  int l16 = lane & 15, hi = lane >> 4;
  int khA = hi * 8, khB = hi * 8;

  v8f acc[4][4] = {};
  for (int kb = 0; kb < Dn; kb += 32) {
    v16bf a[4], b[4];
#pragma unroll
    for (int i = 0; i < 4; ++i) {
      const bf16* rp = W + (size_t)(nbase + i * 16 + l16) * Dn + kb;
      a[i] = make_a(rp + khA, rp + 16 + khA);
    }
#pragma unroll
    for (int j = 0; j < 4; ++j) {
      const bf16* rp = hBF + (size_t)(mbase + j * 16 + l16) * Dn + kb + khB;
      b[j] = make_b(rp);
    }
#pragma unroll
    for (int i = 0; i < 4; ++i)
#pragma unroll
      for (int j = 0; j < 4; ++j)
        acc[i][j] = WMMA_BF16(a[i], b[j], acc[i][j]);
  }
#pragma unroll
  for (int i = 0; i < 4; ++i) {
    int n0 = nbase + i * 16 + 8 * hi;       // 8 consecutive n per lane
    v4f bv0 = *(const v4f*)(bias + n0);
    v4f bv1 = *(const v4f*)(bias + n0 + 4);
#pragma unroll
    for (int j = 0; j < 4; ++j) {
      int m = mbase + j * 16 + l16;
      v8bf o;
#pragma unroll
      for (int r = 0; r < 8; ++r) {
        float bb = (r < 4) ? bv0[r] : bv1[r - 4];
        o[r] = (bf16)(acc[i][j][r] + bb);
      }
      *(v8bf*)(out + (size_t)m * Dn + n0) = o;
    }
  }
}

// ---------------------------------------------------------------------------
// V projection, stored transposed: Vt[b][h][dh][s]  (= column-major per head),
// exactly the B-operand layout the context WMMA needs. Normal orientation:
// lane holds fixed (h,dh), 8 consecutive s -> contiguous 16B stores.
// ---------------------------------------------------------------------------
__global__ __launch_bounds__(256) void proj_v(
    const bf16* __restrict__ hBF,
    const bf16* __restrict__ wvB, const float* __restrict__ bv,
    bf16* __restrict__ Vt) {
  int lane = threadIdx.x & 31;
  int w    = threadIdx.x >> 5;
  int wm = w & 3, wn = w >> 2;
  int mbase = blockIdx.x * 256 + wm * 64;
  int nbase = blockIdx.y * 128 + wn * 64;
  int l16 = lane & 15, hi = lane >> 4;
  int khA = hi * 8, khB = hi * 8;

  v8f acc[4][4] = {};   // [i = m-tile][j = n-tile]
  for (int kb = 0; kb < Dn; kb += 32) {
    v16bf a[4], b[4];
#pragma unroll
    for (int i = 0; i < 4; ++i) {
      const bf16* rp = hBF + (size_t)(mbase + i * 16 + l16) * Dn + kb;
      a[i] = make_a(rp + khA, rp + 16 + khA);
    }
#pragma unroll
    for (int j = 0; j < 4; ++j) {
      const bf16* rp = wvB + (size_t)(nbase + j * 16 + l16) * Dn + kb + khB;
      b[j] = make_b(rp);
    }
#pragma unroll
    for (int i = 0; i < 4; ++i)
#pragma unroll
      for (int j = 0; j < 4; ++j)
        acc[i][j] = WMMA_BF16(a[i], b[j], acc[i][j]);
  }
#pragma unroll
  for (int i = 0; i < 4; ++i) {
    int m0 = mbase + i * 16 + 8 * hi;       // 8 consecutive m (same batch: Sn%16==0)
    int b_ = m0 / Sn;
    int s0 = m0 % Sn;
#pragma unroll
    for (int j = 0; j < 4; ++j) {
      int n  = nbase + j * 16 + l16;
      int h  = n / Dh, dh = n % Dh;
      float bb = bv[n];
      v8bf o;
#pragma unroll
      for (int r = 0; r < 8; ++r) o[r] = (bf16)(acc[i][j][r] + bb);
      *(v8bf*)(Vt + ((size_t)(b_ * Hn + h) * Dh + dh) * Sn + s0) = o;
    }
  }
}

// ---------------------------------------------------------------------------
// cdd attention: block (c,h,b); queries = 64 rows of block c; allowed keys =
// 64 block keys (K proj) + 128 term keys (K proj) = 192. Softmax over the
// 192 allowed keys is exact: masked-out columns hit finfo.min -> exp == 0.
// 4 waves, each owns 16 query rows. In-register softmax on C-fragments,
// P staged through LDS to re-layout into A-operand form.
// ---------------------------------------------------------------------------
__global__ __launch_bounds__(128) void attn_cdd(
    const bf16* __restrict__ Qbf, const bf16* __restrict__ Kbf,
    const bf16* __restrict__ Vt,  const float* __restrict__ am,
    float* __restrict__ outp) {
  constexpr int KT = 12;          // 192 / 16
  constexpr int NK = 192;
  __shared__ bf16 Plds[4][16][NK];

  int c = blockIdx.x, h = blockIdx.y, b = blockIdx.z;
  int lane = threadIdx.x & 31;
  int w    = threadIdx.x >> 5;
  int l16 = lane & 15, hi = lane >> 4;
  int khA = hi * 8, khB = hi * 8;
  int qrow0 = c * Ln + w * 16;    // global s of this wave's first query row

  // ---- scores: 16 x 192, K-dim = Dh
  v8f acc[KT] = {};
  const bf16* Qrow = Qbf + ((size_t)b * Sn + qrow0 + l16) * Dn + h * Dh;
#pragma unroll
  for (int db = 0; db < Dh; db += 32) {
    v16bf aq = make_a(Qrow + db + khA, Qrow + db + 16 + khA);
#pragma unroll
    for (int j = 0; j < KT; ++j) {
      int col = j * 16 + l16;
      int sk  = (col < Ln) ? (c * Ln + col) : (AF + col - Ln);
      const bf16* kp = Kbf + ((size_t)b * Sn + sk) * Dn + h * Dh + db + khB;
      acc[j] = WMMA_BF16(aq, make_b(kp), acc[j]);
    }
  }

  // ---- masked softmax in-register (C-layout: row = r + 8*hi across 16-lane half)
  const float scale = 0.125f;     // 1/sqrt(64)
#pragma unroll
  for (int r = 0; r < 8; ++r) {
    int m = r + 8 * hi;
    size_t mrow = ((size_t)b * Sn + (qrow0 + m)) * Sn;
    float mx = -__builtin_inff();
#pragma unroll
    for (int j = 0; j < KT; ++j) {
      int col = j * 16 + l16;
      int sk  = (col < Ln) ? (c * Ln + col) : (AF + col - Ln);
      float mval = am[mrow + sk];
      float s = (mval != 0.0f) ? acc[j][r] * scale : -__builtin_inff();
      acc[j][r] = s;
      mx = fmaxf(mx, s);
    }
#pragma unroll
    for (int d_ = 1; d_ < 16; d_ <<= 1) mx = fmaxf(mx, __shfl_xor(mx, d_, 32));
    float sm = 0.f;
#pragma unroll
    for (int j = 0; j < KT; ++j) {
      float s = acc[j][r];
      float e = (s == -__builtin_inff()) ? 0.f : __expf(s - mx);
      acc[j][r] = e;
      sm += e;
    }
#pragma unroll
    for (int d_ = 1; d_ < 16; d_ <<= 1) sm += __shfl_xor(sm, d_, 32);
    float inv = (sm > 0.f) ? 1.f / sm : 0.f;   // all-masked row -> zeros (matches ref)
#pragma unroll
    for (int j = 0; j < KT; ++j)
      Plds[w][m][j * 16 + l16] = (bf16)(acc[j][r] * inv);
  }
  __syncthreads();

  // ---- context: 16 x 64 = P(16x192) @ Vset(192x64); B from Vt (column-major)
  v8f ctx[4] = {};
  const bf16* prow = &Plds[w][l16][0];
#pragma unroll
  for (int kt = 0; kt < NK; kt += 32) {
    v16bf ap = make_a(prow + kt + khA, prow + kt + 16 + khA);
    int k0  = kt + khB;           // 16-chunk never straddles the 64-key boundary
    int sk0 = (k0 < Ln) ? (c * Ln + k0) : (AF + k0 - Ln);
#pragma unroll
    for (int dt = 0; dt < 4; ++dt) {
      int d = dt * 16 + l16;
      const bf16* vp = Vt + ((size_t)(b * Hn + h) * Dh + d) * Sn + sk0;
      ctx[dt] = WMMA_BF16(ap, make_b(vp), ctx[dt]);
    }
  }
#pragma unroll
  for (int dt = 0; dt < 4; ++dt) {
    int d = dt * 16 + l16;
#pragma unroll
    for (int r = 0; r < 8; ++r) {
      int q = qrow0 + r + 8 * hi;
      outp[((size_t)b * Sn + q) * Dn + h * Dh + d] = ctx[dt][r];
    }
  }
}

// ---------------------------------------------------------------------------
// pst attention: Q(pst) vs Q(pst) (keys come from the Wq projection!),
// context over V's last T rows. 8 waves x 16 query rows, 128 keys.
// ---------------------------------------------------------------------------
__global__ __launch_bounds__(256) void attn_pst(
    const bf16* __restrict__ Qbf, const bf16* __restrict__ Vt,
    const float* __restrict__ am, float* __restrict__ outp) {
  constexpr int KT = 8;           // 128 / 16
  constexpr int NK = 128;
  __shared__ bf16 Plds[8][16][NK];

  int h = blockIdx.x, b = blockIdx.y;
  int lane = threadIdx.x & 31;
  int w    = threadIdx.x >> 5;
  int l16 = lane & 15, hi = lane >> 4;
  int khA = hi * 8, khB = hi * 8;
  int qrow0 = AF + w * 16;

  v8f acc[KT] = {};
  const bf16* Qrow = Qbf + ((size_t)b * Sn + qrow0 + l16) * Dn + h * Dh;
#pragma unroll
  for (int db = 0; db < Dh; db += 32) {
    v16bf aq = make_a(Qrow + db + khA, Qrow + db + 16 + khA);
#pragma unroll
    for (int j = 0; j < KT; ++j) {
      int sk = AF + j * 16 + l16;
      const bf16* kp = Qbf + ((size_t)b * Sn + sk) * Dn + h * Dh + db + khB;
      acc[j] = WMMA_BF16(aq, make_b(kp), acc[j]);
    }
  }
  const float scale = 0.125f;
#pragma unroll
  for (int r = 0; r < 8; ++r) {
    int m = r + 8 * hi;
    size_t mrow = ((size_t)b * Sn + (qrow0 + m)) * Sn + AF;
    float mx = -__builtin_inff();
#pragma unroll
    for (int j = 0; j < KT; ++j) {
      float mval = am[mrow + j * 16 + l16];
      float s = (mval != 0.0f) ? acc[j][r] * scale : -__builtin_inff();
      acc[j][r] = s;
      mx = fmaxf(mx, s);
    }
#pragma unroll
    for (int d_ = 1; d_ < 16; d_ <<= 1) mx = fmaxf(mx, __shfl_xor(mx, d_, 32));
    float sm = 0.f;
#pragma unroll
    for (int j = 0; j < KT; ++j) {
      float s = acc[j][r];
      float e = (s == -__builtin_inff()) ? 0.f : __expf(s - mx);
      acc[j][r] = e;
      sm += e;
    }
#pragma unroll
    for (int d_ = 1; d_ < 16; d_ <<= 1) sm += __shfl_xor(sm, d_, 32);
    float inv = (sm > 0.f) ? 1.f / sm : 0.f;
#pragma unroll
    for (int j = 0; j < KT; ++j)
      Plds[w][m][j * 16 + l16] = (bf16)(acc[j][r] * inv);
  }
  __syncthreads();

  v8f ctx[4] = {};
  const bf16* prow = &Plds[w][l16][0];
#pragma unroll
  for (int kt = 0; kt < NK; kt += 32) {
    v16bf ap = make_a(prow + kt + khA, prow + kt + 16 + khA);
    int sk0 = AF + kt + khB;
#pragma unroll
    for (int dt = 0; dt < 4; ++dt) {
      int d = dt * 16 + l16;
      const bf16* vp = Vt + ((size_t)(b * Hn + h) * Dh + d) * Sn + sk0;
      ctx[dt] = WMMA_BF16(ap, make_b(vp), ctx[dt]);
    }
  }
#pragma unroll
  for (int dt = 0; dt < 4; ++dt) {
    int d = dt * 16 + l16;
#pragma unroll
    for (int r = 0; r < 8; ++r) {
      int q = qrow0 + r + 8 * hi;
      outp[((size_t)b * Sn + q) * Dn + h * Dh + d] = ctx[dt][r];
    }
  }
}

// ---------------------------------------------------------------------------
extern "C" void kernel_launch(void* const* d_in, const int* in_sizes, int n_in,
                              void* d_out, int out_size, void* d_ws, size_t ws_size,
                              hipStream_t stream) {
  (void)in_sizes; (void)n_in; (void)out_size; (void)ws_size;
  const float* hidden = (const float*)d_in[0];
  const float* am     = (const float*)d_in[1];
  const float* Wq = (const float*)d_in[2];
  const float* bq = (const float*)d_in[3];
  const float* Wk = (const float*)d_in[4];
  const float* bk = (const float*)d_in[5];
  const float* Wv = (const float*)d_in[6];
  const float* bv = (const float*)d_in[7];
  float* out = (float*)d_out;

  char* ws = (char*)d_ws;
  size_t off = 0;
  auto take = [&](size_t bytes) -> char* {
    char* p = ws + off;
    off += (bytes + 255) & ~(size_t)255;
    return p;
  };
  bf16* hBF = (bf16*)take((size_t)Mn * Dn * sizeof(bf16));
  bf16* wqB = (bf16*)take((size_t)Dn * Dn * sizeof(bf16));
  bf16* wkB = (bf16*)take((size_t)Dn * Dn * sizeof(bf16));
  bf16* wvB = (bf16*)take((size_t)Dn * Dn * sizeof(bf16));
  bf16* Qbf = (bf16*)take((size_t)Mn * Dn * sizeof(bf16));
  bf16* Kbf = (bf16*)take((size_t)Mn * Dn * sizeof(bf16));
  bf16* Vt  = (bf16*)take((size_t)Mn * Dn * sizeof(bf16));

  const int nh = Mn * Dn, nw = Dn * Dn;
  f32_to_bf16<<<(nh / 8 + 255) / 256, 256, 0, stream>>>(hidden, hBF, nh);
  f32_to_bf16<<<(nw / 8 + 255) / 256, 256, 0, stream>>>(Wq, wqB, nw);
  f32_to_bf16<<<(nw / 8 + 255) / 256, 256, 0, stream>>>(Wk, wkB, nw);
  f32_to_bf16<<<(nw / 8 + 255) / 256, 256, 0, stream>>>(Wv, wvB, nw);

  proj_qk<<<dim3(Mn / 256, Dn / 128, 2), 256, 0, stream>>>(hBF, wqB, bq, wkB, bk, Qbf, Kbf);
  proj_v <<<dim3(Mn / 256, Dn / 128),    256, 0, stream>>>(hBF, wvB, bv, Vt);

  attn_cdd<<<dim3(CDD, Hn, Bn), 128, 0, stream>>>(Qbf, Kbf, Vt, am, out);
  attn_pst<<<dim3(Hn, Bn),      256, 0, stream>>>(Qbf, Vt, am, out);
}
